// GCNConv_28080496181834
// MI455X (gfx1250) — compile-verified, matrix operand order checked by reading, not run on hardware
//
#include <hip/hip_runtime.h>
#include <hip/hip_bf16.h>

// ---------------------------------------------------------------------------
// GCNConv on MI455X (gfx1250, wave32).
//
// h = x @ W.T runs on the WMMA unit via v_wmma_f32_16x16x32_bf16 using a
// split-bf16 (hi/lo) decomposition for ~fp32 accuracy:
//   x*w ~= x_hi*w_hi + x_hi*w_lo + x_lo*w_hi   (3 bf16 WMMAs per k-step/tile)
//
// W is pre-packed once per call into the exact per-lane B-matrix register
// layout (hi/lo planes) so the GEMM inner loop loads B with plain b128 loads
// and spends no VALU on W conversion. Each wave computes a 16x64 strip
// (4 C tiles) so one A hi/lo split feeds 12 WMMAs.
//
// Degree / scatter phases are the memory-roofline part (~1.7GB at 23.3TB/s)
// and use coalesced float4 gathers + global_atomic_add_f32 scatters.
// ---------------------------------------------------------------------------

typedef __bf16 bf16_t;
typedef bf16_t v16bf __attribute__((ext_vector_type(16)));
typedef float  v8f   __attribute__((ext_vector_type(8)));

#define D_DIM 256
#define KSTEPS 8              // D / 32
#define COL_TILES 16          // D / 16
#define COLS_PER_WAVE 4       // C tiles per wave (16x64 strip)

// Per-lane K-pair base offset inside a 16x32 bf16 A/B tile (ISA 7.12.2).
__device__ __forceinline__ int kpair_base(int r, int half) {
    return (r < 4) ? (8 * half + 2 * r) : (16 + 8 * half + 2 * (r - 4));
}

// ---- phase 1: degree init (self-loop contributes 1 to every node) ----------
__global__ void deg_init_kernel(float* __restrict__ deg, int N) {
    int i = blockIdx.x * blockDim.x + threadIdx.x;
    if (i < N) deg[i] = 1.0f;
}

// ---- phase 2: deg[dst] += 1 per edge ---------------------------------------
__global__ void deg_accum_kernel(const long long* __restrict__ dst,
                                 float* __restrict__ deg, int E) {
    int e = blockIdx.x * blockDim.x + threadIdx.x;
    if (e < E) atomicAdd(&deg[(int)dst[e]], 1.0f);
}

// ---- phase 3: dinv = rsqrt(deg) in place (deg >= 1 always) -----------------
__global__ void dinv_kernel(float* __restrict__ deg, int N) {
    int i = blockIdx.x * blockDim.x + threadIdx.x;
    if (i < N) deg[i] = rsqrtf(deg[i]);
}

// ---- phase 4a: pre-pack W into per-lane B-matrix layout, hi/lo planes ------
// wpk index = ((colTile*KSTEPS + t)*2 + sel)*32 + lane, each entry = v16bf.
// B column n of tile (c,t) is W row c*16+n, k range [t*32, t*32+32).
__global__ void __launch_bounds__(256)
wpack_kernel(const float* __restrict__ W, v16bf* __restrict__ wpk) {
    const int lane = threadIdx.x & 31;
    const int wave = (blockIdx.x * blockDim.x + threadIdx.x) >> 5;  // 0..127
    const int c = wave >> 3;          // colTile 0..15
    const int t = wave & 7;           // kstep  0..7
    const int n = lane & 15, half = lane >> 4;

    const float* __restrict__ wrow = W + (size_t)(c * 16 + n) * D_DIM + t * 32;
    v16bf hi, lo;
#pragma unroll
    for (int r = 0; r < 8; ++r) {
        const int kb = kpair_base(r, half);
        const float2 v = *(const float2*)(wrow + kb);
        bf16_t h0 = (bf16_t)v.x;  bf16_t h1 = (bf16_t)v.y;
        hi[2 * r]     = h0;  hi[2 * r + 1] = h1;
        lo[2 * r]     = (bf16_t)(v.x - (float)h0);
        lo[2 * r + 1] = (bf16_t)(v.y - (float)h1);
    }
    const size_t base = ((size_t)(c * KSTEPS + t) * 2) * 32 + lane;
    wpk[base]      = hi;   // sel 0
    wpk[base + 32] = lo;   // sel 1
}

// ---- phase 4b: h = x @ W.T via WMMA, 16x64 strip per wave ------------------
__global__ void __launch_bounds__(256)
gemm_wmma_kernel(const float* __restrict__ x, const v16bf* __restrict__ wpk,
                 float* __restrict__ h, int N) {
    const int lane = threadIdx.x & 31;
    const int wave = (blockIdx.x * blockDim.x + threadIdx.x) >> 5;
    const int colGroups = COL_TILES / COLS_PER_WAVE;   // 4
    const int rowTile = wave / colGroups;
    const int cg      = wave % colGroups;
    if (rowTile * 16 >= N) return;                     // wave-uniform predicate

    const int mn   = lane & 15;
    const int half = lane >> 4;

    const float* __restrict__ xrow = x + (size_t)(rowTile * 16 + mn) * D_DIM;

    v8f acc[COLS_PER_WAVE] = {};
    for (int t = 0; t < KSTEPS; ++t) {
        const int k0 = t * 32;

        // Split A (x tile) into bf16 hi/lo once; reused by all 4 column tiles.
        v16bf ahi, alo;
#pragma unroll
        for (int r = 0; r < 8; ++r) {
            const int kb = kpair_base(r, half);
            const float2 av = *(const float2*)(xrow + k0 + kb);
            bf16_t a0 = (bf16_t)av.x;  bf16_t a1 = (bf16_t)av.y;
            ahi[2 * r]     = a0;  ahi[2 * r + 1] = a1;
            alo[2 * r]     = (bf16_t)(av.x - (float)a0);
            alo[2 * r + 1] = (bf16_t)(av.y - (float)a1);
        }

#pragma unroll
        for (int j = 0; j < COLS_PER_WAVE; ++j) {
            const int c = cg * COLS_PER_WAVE + j;
            const size_t bidx = ((size_t)(c * KSTEPS + t) * 2) * 32 + lane;
            const v16bf bhi = wpk[bidx];
            const v16bf blo = wpk[bidx + 32];
            acc[j] = __builtin_amdgcn_wmma_f32_16x16x32_bf16(
                false, ahi, false, bhi, (short)0, acc[j], false, false);
            acc[j] = __builtin_amdgcn_wmma_f32_16x16x32_bf16(
                false, ahi, false, blo, (short)0, acc[j], false, false);
            acc[j] = __builtin_amdgcn_wmma_f32_16x16x32_bf16(
                false, alo, false, bhi, (short)0, acc[j], false, false);
        }
    }

    // Store 4 tiles: h[rowTile*16 + m][cg*64 + j*16 + n]
    float* __restrict__ hbase =
        h + (size_t)(rowTile * 16) * D_DIM + cg * (COLS_PER_WAVE * 16) + mn;
#pragma unroll
    for (int j = 0; j < COLS_PER_WAVE; ++j) {
#pragma unroll
        for (int r = 0; r < 8; ++r) {
            const int m = r + 8 * half;
            hbase[(size_t)m * D_DIM + j * 16] = acc[j][r];
        }
    }
}

// ---- phase 5: out = a*x + (1-a)*(b + dinv[i]^2 * h[i]) (self-loop term) ----
__global__ void out_init_kernel(const float* __restrict__ x,
                                const float* __restrict__ h,
                                const float* __restrict__ dinv,
                                const float* __restrict__ b,
                                const float* __restrict__ alpha,
                                float* __restrict__ out, int N) {
    long long idx = (long long)blockIdx.x * blockDim.x + threadIdx.x;
    long long total = (long long)N * D_DIM;
    if (idx >= total) return;
    const int i = (int)(idx >> 8);     // / 256
    const int d = (int)(idx & 255);
    const float a  = alpha[0];
    const float di = dinv[i];
    out[idx] = a * x[idx] + (1.0f - a) * (b[d] + di * di * h[idx]);
}

// ---- phase 6: out[dst] += (1-a)*dinv[s]*dinv[d]*h[src] over edges ----------
__global__ void scatter_kernel(const long long* __restrict__ src,
                               const long long* __restrict__ dst,
                               const float* __restrict__ h,
                               const float* __restrict__ dinv,
                               const float* __restrict__ alpha,
                               float* __restrict__ out, int E) {
    long long t = (long long)blockIdx.x * blockDim.x + threadIdx.x;
    int e = (int)(t >> 6);
    if (e >= E) return;
    const int d = ((int)t & 63) * 4;
    const int s = (int)src[e];
    const int g = (int)dst[e];
    const float scale = (1.0f - alpha[0]) * dinv[s] * dinv[g];

    const float4 hv = *(const float4*)(h + (size_t)s * D_DIM + d);
    float* __restrict__ o = out + (size_t)g * D_DIM + d;
    atomicAdd(o + 0, scale * hv.x);
    atomicAdd(o + 1, scale * hv.y);
    atomicAdd(o + 2, scale * hv.z);
    atomicAdd(o + 3, scale * hv.w);
}

// ---------------------------------------------------------------------------
extern "C" void kernel_launch(void* const* d_in, const int* in_sizes, int n_in,
                              void* d_out, int out_size, void* d_ws, size_t ws_size,
                              hipStream_t stream) {
    const float*     x     = (const float*)d_in[0];       // [N, 256]
    const long long* edges = (const long long*)d_in[1];   // [2, E] int64
    const float*     W     = (const float*)d_in[2];       // [256, 256]
    const float*     b     = (const float*)d_in[3];       // [256]
    const float*     alpha = (const float*)d_in[4];       // [1]

    const int N = in_sizes[0] / D_DIM;                    // 50000
    const int E = in_sizes[1] / 2;                        // 800000
    const long long* src = edges;
    const long long* dst = edges + E;

    // workspace: h [N*256 f32] | dinv [N f32] | wpk [16*8*2*32 v16bf = 256KB]
    float* h    = (float*)d_ws;
    float* dinv = h + (size_t)N * D_DIM;
    v16bf* wpk  = (v16bf*)(dinv + N);   // offset is 32B-aligned (N mult of 32)

    float* out = (float*)d_out;

    // 1) deg = 1 (self loops)
    deg_init_kernel<<<(N + 255) / 256, 256, 0, stream>>>(dinv, N);
    // 2) deg[dst] += 1
    deg_accum_kernel<<<(E + 255) / 256, 256, 0, stream>>>(dst, dinv, E);
    // 3) dinv = rsqrt(deg)
    dinv_kernel<<<(N + 255) / 256, 256, 0, stream>>>(dinv, N);
    // 4a) pre-pack W into B-register layout (hi/lo): 128 waves
    wpack_kernel<<<16, 256, 0, stream>>>(W, wpk);
    // 4b) h = x @ W.T  (WMMA; 16x64 strip per wave, 8 waves per block)
    {
        const int waves  = (N / 16) * (COL_TILES / COLS_PER_WAVE);  // 12500
        const int blocks = (waves + 7) / 8;                         // 1563
        gemm_wmma_kernel<<<blocks, 256, 0, stream>>>(x, wpk, h, N);
    }
    // 5) out = a*x + (1-a)*(b + dinv^2 * h)
    {
        const long long total = (long long)N * D_DIM;
        const int blocks = (int)((total + 255) / 256);
        out_init_kernel<<<blocks, 256, 0, stream>>>(x, h, dinv, b, alpha, out, N);
    }
    // 6) edge scatter-add
    {
        const long long threads = (long long)E * 64;
        const int blocks = (int)((threads + 255) / 256);  // 200000
        scatter_kernel<<<blocks, 256, 0, stream>>>(src, dst, h, dinv, alpha, out, E);
    }
}
